// Decoder_5858335391999
// MI455X (gfx1250) — matrix operand order repaired and checked by reading, not compiled, not run on hardware
//
#include <hip/hip_runtime.h>
#include <hip/hip_bf16.h>
#include <math.h>

// Shapes from the reference
#define B_ 64
#define H_ 512
#define S_ 128
#define T_ 64
#define G4 2048   // 4*H

typedef __attribute__((ext_vector_type(16))) __bf16 v16bf;
typedef __attribute__((ext_vector_type(8)))  __bf16 v8bf;
typedef __attribute__((ext_vector_type(8)))  float  v8f;

__device__ __forceinline__ unsigned short f2bf_bits(float x) {
  unsigned int u = __float_as_uint(x);
  u += 0x7fffu + ((u >> 16) & 1u);   // round-to-nearest-even
  return (unsigned short)(u >> 16);
}
__device__ __forceinline__ float sigm(float x) { return 1.0f / (1.0f + __expf(-x)); }

// B fragment (K x 16 tile of W^T): lane = column (lane&15), K-half via lane>>4;
// 16 contiguous bf16 from a row of W[N,K] -> one 32B load.
__device__ __forceinline__ v16bf load_bfrag(const unsigned short* p) {
  return *(const v16bf*)p;
}
// A fragment (16 x 32 tile): lane row = lane&15; 16 contiguous K at 16*(lane>>4).
__device__ __forceinline__ v16bf load_afrag(const unsigned short* p) {
  v8bf lo = *(const v8bf*)p;
  v8bf hi = *(const v8bf*)(p + 8);
  return __builtin_shufflevector(lo, hi, 0,1,2,3,4,5,6,7,8,9,10,11,12,13,14,15);
}
__device__ __forceinline__ v8f wmma_bf16(v16bf a, v16bf b, v8f c) {
  return __builtin_amdgcn_wmma_f32_16x16x32_bf16(false, a, false, b, (short)0, c,
                                                 false, false);
}

// ---------------- setup kernels ----------------
__global__ void k_f32_to_bf16(const float* __restrict__ src,
                              unsigned short* __restrict__ dst, int n) {
  int i = blockIdx.x * blockDim.x + threadIdx.x;
  if (i < n) dst[i] = f2bf_bits(src[i]);
}

__global__ void k_zero_bf(unsigned short* __restrict__ dst, int n) {
  int i = blockIdx.x * blockDim.x + threadIdx.x;
  if (i < n) dst[i] = 0;   // bf16 +0.0
}

__global__ void k_init_state(const float* __restrict__ h0, const float* __restrict__ c0,
                             float* __restrict__ hstate_f, unsigned short* __restrict__ hstate_bf,
                             float* __restrict__ cstate) {
  int i = blockIdx.x * blockDim.x + threadIdx.x;   // n = 2*B*H (parity-0 block)
  if (i < 2 * B_ * H_) {
    float h = h0[i];
    hstate_f[i] = h;
    hstate_bf[i] = f2bf_bits(h);
    cstate[i] = c0[i];
  }
}

__global__ void k_final_copy(const float* __restrict__ hstate_f0,
                             const float* __restrict__ cstate0,
                             float* __restrict__ out_h, float* __restrict__ out_c) {
  int i = blockIdx.x * blockDim.x + threadIdx.x;   // n = 2*B*H
  if (i < 2 * B_ * H_) { out_h[i] = hstate_f0[i]; out_c[i] = cstate0[i]; }
}

// Gather this step's embedding rows into emb_bf[64,512].
__global__ void k_embed_gather(const int* __restrict__ tok,
                               const float* __restrict__ emb,
                               unsigned short* __restrict__ ebf) {
  int b = blockIdx.x;
  int t = tok[b];
  const float* row = emb + (size_t)t * 512;
  for (int e = threadIdx.x; e < 512; e += blockDim.x)
    ebf[b * 512 + e] = f2bf_bits(row[e]);
}

// ---------------- fused LSTM layer: WMMA gates + cell update ----------------
// grid(32) = gate-col tiles, block(128) = 4 waves = 4 M tiles.
// A has up to two K=512 input segments (X0, optional X1) plus the H segment; W
// row offsets follow the concat order. Each wave computes the i/f/g/o tiles for
// one (m,n) tile sharing A fragments, then applies the cell nonlinearity.
__global__ void k_lstm_layer(const unsigned short* __restrict__ X0,   // [64,512]
                             const unsigned short* __restrict__ X1,   // [64,512] or nullptr
                             const unsigned short* __restrict__ Hbf,  // [64,512]
                             const unsigned short* __restrict__ Wih,  // [2048, KW]
                             int KW,                                  // 1024 or 512
                             const unsigned short* __restrict__ Whh,  // [2048, 512]
                             const float* __restrict__ bih, const float* __restrict__ bhh,
                             const float* __restrict__ Cin,
                             float* __restrict__ Cout,
                             float* __restrict__ Hf_out,
                             unsigned short* __restrict__ Hbf_out) {
  const int lane = threadIdx.x & 31;
  const int mt = threadIdx.x >> 5;      // m tile 0..3
  const int nt = blockIdx.x;            // gate-col tile 0..31
  const int l16 = lane & 15;
  const int kh  = lane >> 4;
  const int arow = mt * 16 + l16;
  const int bcol = nt * 16 + l16;       // hidden index within a gate

  const unsigned short* w0 = Wih + (size_t)(0 * 512 + bcol) * KW;
  const unsigned short* w1 = Wih + (size_t)(1 * 512 + bcol) * KW;
  const unsigned short* w2 = Wih + (size_t)(2 * 512 + bcol) * KW;
  const unsigned short* w3 = Wih + (size_t)(3 * 512 + bcol) * KW;

  v8f a0 = {0,0,0,0,0,0,0,0}, a1 = a0, a2 = a0, a3 = a0;

  // segment 0: X0 (W cols 0..511)
  for (int kb = 0; kb < 512; kb += 32) {
    v16bf a = load_afrag(X0 + arow * 512 + kb + 16 * kh);
    int ko = kb + 16 * kh;
    a0 = wmma_bf16(a, load_bfrag(w0 + ko), a0);
    a1 = wmma_bf16(a, load_bfrag(w1 + ko), a1);
    a2 = wmma_bf16(a, load_bfrag(w2 + ko), a2);
    a3 = wmma_bf16(a, load_bfrag(w3 + ko), a3);
  }
  // segment 1 (uniform branch): X1 (W cols 512..1023)
  if (X1) {
    for (int kb = 0; kb < 512; kb += 32) {
      v16bf a = load_afrag(X1 + arow * 512 + kb + 16 * kh);
      int ko = 512 + kb + 16 * kh;
      a0 = wmma_bf16(a, load_bfrag(w0 + ko), a0);
      a1 = wmma_bf16(a, load_bfrag(w1 + ko), a1);
      a2 = wmma_bf16(a, load_bfrag(w2 + ko), a2);
      a3 = wmma_bf16(a, load_bfrag(w3 + ko), a3);
    }
  }
  // recurrent segment: H @ Whh^T
  const unsigned short* u0 = Whh + (size_t)(0 * 512 + bcol) * 512;
  const unsigned short* u1 = Whh + (size_t)(1 * 512 + bcol) * 512;
  const unsigned short* u2 = Whh + (size_t)(2 * 512 + bcol) * 512;
  const unsigned short* u3 = Whh + (size_t)(3 * 512 + bcol) * 512;
  for (int kb = 0; kb < 512; kb += 32) {
    v16bf a = load_afrag(Hbf + arow * 512 + kb + 16 * kh);
    int ko = kb + 16 * kh;
    a0 = wmma_bf16(a, load_bfrag(u0 + ko), a0);
    a1 = wmma_bf16(a, load_bfrag(u1 + ko), a1);
    a2 = wmma_bf16(a, load_bfrag(u2 + ko), a2);
    a3 = wmma_bf16(a, load_bfrag(u3 + ko), a3);
  }

  // all 8 acc elements of a lane share one column -> one bias per gate
  const float bi = bih[0 * 512 + bcol] + bhh[0 * 512 + bcol];
  const float bf = bih[1 * 512 + bcol] + bhh[1 * 512 + bcol];
  const float bg = bih[2 * 512 + bcol] + bhh[2 * 512 + bcol];
  const float bo = bih[3 * 512 + bcol] + bhh[3 * 512 + bcol];

#pragma unroll
  for (int v = 0; v < 8; ++v) {
    int m = mt * 16 + v + 8 * kh;       // batch row of this D element
    float gi = a0[v] + bi;
    float gf = a1[v] + bf;
    float gg = a2[v] + bg;
    float go = a3[v] + bo;
    float co = Cin[m * 512 + bcol];
    float cn = sigm(gf) * co + sigm(gi) * tanhf(gg);
    float hn = sigm(go) * tanhf(cn);
    Cout[m * 512 + bcol] = cn;
    Hf_out[m * 512 + bcol] = hn;
    Hbf_out[m * 512 + bcol] = f2bf_bits(hn);
  }
}

// ---------------- q = h1 @ w_in^T via WMMA ([64,512]x[512,512]) ---------------
__global__ void k_q_wmma(const unsigned short* __restrict__ Hbf,   // [64,512]
                         const unsigned short* __restrict__ Win,   // [512,512]
                         float* __restrict__ qout) {               // [64,512]
  const int lane = threadIdx.x & 31;
  const int mt = threadIdx.x >> 5;
  const int nt = blockIdx.x;            // 0..31
  const int l16 = lane & 15;
  const int kh  = lane >> 4;
  const int arow = mt * 16 + l16;
  const unsigned short* wr = Win + (size_t)(nt * 16 + l16) * 512;
  v8f acc = {0,0,0,0,0,0,0,0};
  for (int kb = 0; kb < 512; kb += 32) {
    v16bf a = load_afrag(Hbf + arow * 512 + kb + 16 * kh);
    v16bf b = load_bfrag(wr + kb + 16 * kh);
    acc = wmma_bf16(a, b, acc);
  }
  const int col = nt * 16 + l16;
#pragma unroll
  for (int v = 0; v < 8; ++v)
    qout[(mt * 16 + v + 8 * kh) * 512 + col] = acc[v];
}

// ------- attention: masked softmax + ctx (genuinely per-batch -> VALU) --------
__global__ void k_attention(const float* __restrict__ qg,       // [B,H] from k_q_wmma
                            const float* __restrict__ mb,       // [B,S,H]
                            const int* __restrict__ lengths,
                            float* __restrict__ attn_p,         // d_out attns + t*B*S
                            unsigned short* __restrict__ ctx_bf) { // [B,H]
  __shared__ float q[512];
  __shared__ float sc[128];
  const int b = blockIdx.x;
  const int tid = threadIdx.x;          // 128 threads
  for (int j = tid; j < 512; j += 128) q[j] = qg[b * 512 + j];
  __syncthreads();
  {
    const float* mrow = mb + ((size_t)b * 128 + tid) * 512;
    float s = 0.f;
    for (int k = 0; k < 512; ++k) s = fmaf(q[k], mrow[k], s);
    sc[tid] = (tid < lengths[b]) ? s : -INFINITY;
  }
  __syncthreads();
  float mx = -INFINITY;
  for (int s = 0; s < 128; ++s) mx = fmaxf(mx, sc[s]);
  float e = __expf(sc[tid] - mx);       // masked -> exp(-inf) = 0
  __syncthreads();
  sc[tid] = e;
  __syncthreads();
  float sum = 0.f;
  for (int s = 0; s < 128; ++s) sum += sc[s];
  float p = e / sum;
  attn_p[b * 128 + tid] = p;
  __syncthreads();
  sc[tid] = p;
  __syncthreads();
  for (int j = tid; j < 512; j += 128) {
    float c = 0.f;
    for (int s = 0; s < 128; ++s) c = fmaf(sc[s], mb[((size_t)b * 128 + s) * 512 + j], c);
    ctx_bf[b * 512 + j] = f2bf_bits(c);
  }
}

// ------- attn_h = tanh([ctx,h1] @ w_out^T) via WMMA; segmented A --------------
// Writes dec_outs[t] (f32) and next step's feed_bf directly.
__global__ void k_attn_out(const unsigned short* __restrict__ Cbf,   // ctx [64,512]
                           const unsigned short* __restrict__ H1bf,  // h1  [64,512]
                           const unsigned short* __restrict__ Wout,  // [512,1024]
                           float* __restrict__ dec_out,
                           unsigned short* __restrict__ feed_bf) {   // [64,512]
  const int lane = threadIdx.x & 31;
  const int mt = threadIdx.x >> 5;
  const int nt = blockIdx.x;            // 0..31
  const int l16 = lane & 15;
  const int kh  = lane >> 4;
  const int arow = mt * 16 + l16;
  const unsigned short* wr = Wout + (size_t)(nt * 16 + l16) * 1024;
  v8f acc = {0,0,0,0,0,0,0,0};
  for (int kb = 0; kb < 512; kb += 32) {               // ctx segment (W cols 0..511)
    v16bf a = load_afrag(Cbf + arow * 512 + kb + 16 * kh);
    v16bf b = load_bfrag(wr + kb + 16 * kh);
    acc = wmma_bf16(a, b, acc);
  }
  for (int kb = 0; kb < 512; kb += 32) {               // h1 segment (W cols 512..1023)
    v16bf a = load_afrag(H1bf + arow * 512 + kb + 16 * kh);
    v16bf b = load_bfrag(wr + 512 + kb + 16 * kh);
    acc = wmma_bf16(a, b, acc);
  }
  const int col = nt * 16 + l16;
#pragma unroll
  for (int v = 0; v < 8; ++v) {
    int m = mt * 16 + v + 8 * kh;
    float val = tanhf(acc[v]);
    dec_out[m * 512 + col] = val;
    feed_bf[m * 512 + col] = f2bf_bits(val);
  }
}

// ---------------- host orchestration ----------------
extern "C" void kernel_launch(void* const* d_in, const int* in_sizes, int n_in,
                              void* d_out, int out_size, void* d_ws, size_t ws_size,
                              hipStream_t stream) {
  const int*   tokens    = (const int*)  d_in[0];   // [T,B]
  const float* memory    = (const float*)d_in[1];   // [B,S,H]
  const int*   lengths   = (const int*)  d_in[2];
  const float* h0        = (const float*)d_in[3];   // [2,B,H]
  const float* c0        = (const float*)d_in[4];
  const float* embedding = (const float*)d_in[5];   // [V,E]
  const float* w_ih0     = (const float*)d_in[6];   // [2048,1024]
  const float* w_hh0     = (const float*)d_in[7];   // [2048,512]
  const float* b_ih0     = (const float*)d_in[8];
  const float* b_hh0     = (const float*)d_in[9];
  const float* w_ih1     = (const float*)d_in[10];  // [2048,512]
  const float* w_hh1     = (const float*)d_in[11];
  const float* b_ih1     = (const float*)d_in[12];
  const float* b_hh1     = (const float*)d_in[13];
  const float* w_in      = (const float*)d_in[14];  // [512,512]
  const float* w_out     = (const float*)d_in[15];  // [512,1024]
  float* out = (float*)d_out;

  // workspace carve-up (256B aligned)
  char* ws = (char*)d_ws;
  size_t off = 0;
  auto carve = [&](size_t bytes) -> void* {
    void* p = ws + off;
    off += (bytes + 255) & ~(size_t)255;
    return p;
  };
  unsigned short* wih0_bf = (unsigned short*)carve((size_t)G4 * 1024 * 2);
  unsigned short* whh0_bf = (unsigned short*)carve((size_t)G4 * 512 * 2);
  unsigned short* wih1_bf = (unsigned short*)carve((size_t)G4 * 512 * 2);
  unsigned short* whh1_bf = (unsigned short*)carve((size_t)G4 * 512 * 2);
  unsigned short* win_bf  = (unsigned short*)carve((size_t)512 * 512 * 2);
  unsigned short* wout_bf = (unsigned short*)carve((size_t)512 * 1024 * 2);
  unsigned short* emb_bf  = (unsigned short*)carve((size_t)B_ * H_ * 2);
  unsigned short* feed_bf = (unsigned short*)carve((size_t)B_ * H_ * 2);
  unsigned short* ctx_bf  = (unsigned short*)carve((size_t)B_ * H_ * 2);
  float*          q_f     = (float*)carve((size_t)B_ * H_ * 4);
  unsigned short* hst_bf  = (unsigned short*)carve((size_t)2 * 2 * B_ * H_ * 2); // [par][layer][B][H]
  float*          hst_f   = (float*)carve((size_t)2 * 2 * B_ * H_ * 4);
  float*          cst_f   = (float*)carve((size_t)2 * 2 * B_ * H_ * 4);

  const int LBH = B_ * H_;  // 32768 elements per (parity,layer) block

  // convert weights to bf16 (every call: deterministic, cheap vs total work)
  auto cvt = [&](const float* s, unsigned short* d, int n) {
    k_f32_to_bf16<<<(n + 255) / 256, 256, 0, stream>>>(s, d, n);
  };
  cvt(w_ih0, wih0_bf, G4 * 1024);
  cvt(w_hh0, whh0_bf, G4 * 512);
  cvt(w_ih1, wih1_bf, G4 * 512);
  cvt(w_hh1, whh1_bf, G4 * 512);
  cvt(w_in,  win_bf,  512 * 512);
  cvt(w_out, wout_bf, 512 * 1024);

  k_init_state<<<(2 * LBH + 255) / 256, 256, 0, stream>>>(h0, c0, hst_f, hst_bf, cst_f);
  k_zero_bf<<<(LBH + 255) / 256, 256, 0, stream>>>(feed_bf, LBH);   // feed0 = 0

  const size_t DEC = (size_t)T_ * B_ * H_;             // 2097152
  const size_t ATT = DEC + (size_t)T_ * B_ * S_;       // +524288
  float* out_hf = out + ATT;
  float* out_cf = out + ATT + 2 * LBH;

  for (int t = 0; t < T_; ++t) {
    const int rp = t & 1, wp = rp ^ 1;
    k_embed_gather<<<B_, 256, 0, stream>>>(tokens + t * B_, embedding, emb_bf);
    // layer 0: x = [emb | feed] (K=1024)
    k_lstm_layer<<<32, 128, 0, stream>>>(
        emb_bf, feed_bf,
        hst_bf + (rp * 2 + 0) * LBH,
        wih0_bf, 1024, whh0_bf, b_ih0, b_hh0,
        cst_f + (rp * 2 + 0) * LBH, cst_f + (wp * 2 + 0) * LBH,
        hst_f + (wp * 2 + 0) * LBH, hst_bf + (wp * 2 + 0) * LBH);
    // layer 1: x = h0_new (K=512)
    k_lstm_layer<<<32, 128, 0, stream>>>(
        hst_bf + (wp * 2 + 0) * LBH, nullptr,
        hst_bf + (rp * 2 + 1) * LBH,
        wih1_bf, 512, whh1_bf, b_ih1, b_hh1,
        cst_f + (rp * 2 + 1) * LBH, cst_f + (wp * 2 + 1) * LBH,
        hst_f + (wp * 2 + 1) * LBH, hst_bf + (wp * 2 + 1) * LBH);
    // q = h1_new @ w_in^T (WMMA)
    k_q_wmma<<<32, 128, 0, stream>>>(hst_bf + (wp * 2 + 1) * LBH, win_bf, q_f);
    // masked softmax + ctx (per-batch matvecs, VALU)
    k_attention<<<B_, 128, 0, stream>>>(
        q_f, memory, lengths, out + DEC + (size_t)t * B_ * S_, ctx_bf);
    // attn_h = tanh([ctx|h1] @ w_out^T): dec_outs[t] + next feed (WMMA)
    k_attn_out<<<32, 128, 0, stream>>>(
        ctx_bf, hst_bf + (wp * 2 + 1) * LBH, wout_bf,
        out + (size_t)t * B_ * H_, feed_bf);
  }
  // after step t=63: write parity = 0 -> final state lives in the parity-0 block
  k_final_copy<<<(2 * LBH + 255) / 256, 256, 0, stream>>>(hst_f, cst_f, out_hf, out_cf);
}
